// SelfAttention_34711925686873
// MI455X (gfx1250) — matrix operand order repaired and checked by reading, not compiled, not run on hardware
//
#include <hip/hip_runtime.h>
#include <cstdint>

#define D_MODEL 1024
#define NHEAD 16
#define DEPTH 64
#define SEQ 2048
#define BATCH 2
#define ROWS (BATCH * SEQ) /* 4096 */

#define BM 64
#define BN 64
#define BK 32

typedef __attribute__((ext_vector_type(16))) __bf16 v16bf;
typedef __attribute__((ext_vector_type(8))) float v8f;

union Frag16 {
  v16bf v;
  uint4 q[2];
  unsigned short u[16];
};

// Packed f32->bf16 conversion (RNE) on the VALU: one op per two floats.
__device__ __forceinline__ unsigned pk2(float lo, float hi) {
  unsigned r;
  asm("v_cvt_pk_bf16_f32 %0, %1, %2" : "=v"(r) : "v"(lo), "v"(hi));
  return r;
}
__device__ __forceinline__ unsigned short f2bf(float f) {
  return (unsigned short)pk2(f, f);
}

__device__ __forceinline__ v8f wmma_bf16(const Frag16& a, const Frag16& b, v8f c) {
  return __builtin_amdgcn_wmma_f32_16x16x32_bf16(
      /*neg_a=*/false, a.v, /*neg_b=*/false, b.v,
      /*c_mod=*/(short)0, c, /*reuse_a=*/false, /*reuse_b=*/false);
}

// ---------------------------------------------------------------------------
// Kernel 1: Y = X * W + b  (X f32 [4096,1024], W f32 [1024,1024]).
// VT==0: write bf16 head-major [B,H,S,DEPTH]
// VT==1: write bf16 head-major transposed [B,H,DEPTH,S]  (for V)
// ---------------------------------------------------------------------------
template <int VT>
__global__ __launch_bounds__(128) void qkv_gemm_bf16(
    const float* __restrict__ X, const float* __restrict__ W,
    const float* __restrict__ bias, unsigned short* __restrict__ out_hm) {
  __shared__ __align__(16) unsigned short Alds[BM][BK + 8];
  __shared__ __align__(16) unsigned short Blds[BN][BK + 8]; // transposed [n][k]
  const int tid = threadIdx.x, wave = tid >> 5, lane = tid & 31;
  const int h16 = lane >> 4, l16 = lane & 15;
  const int row0 = blockIdx.x * BM, col0 = blockIdx.y * BN;
  const int ar = tid >> 1, ac = (tid & 1) * 16; // A fill: 16-elem chunk
  v8f acc[4] = {};

  for (int k0 = 0; k0 < D_MODEL; k0 += BK) {
    { // A tile: 64x32 f32 -> packed bf16, b128 traffic only.
      const float4* Ag =
          (const float4*)&X[(size_t)(row0 + ar) * D_MODEL + k0 + ac];
      const float4 f0 = Ag[0], f1 = Ag[1], f2 = Ag[2], f3 = Ag[3];
      *(uint4*)&Alds[ar][ac] = make_uint4(pk2(f0.x, f0.y), pk2(f0.z, f0.w),
                                          pk2(f1.x, f1.y), pk2(f1.z, f1.w));
      *(uint4*)&Alds[ar][ac + 8] = make_uint4(pk2(f2.x, f2.y), pk2(f2.z, f2.w),
                                              pk2(f3.x, f3.y), pk2(f3.z, f3.w));
    }
    // B tile: 32x64 f32, transposed into Blds[n][k]; each thread packs an
    // 8-deep k-strip of one column into a single b128 LDS store.
#pragma unroll
    for (int s = tid; s < 4 * BN; s += 128) {
      const int n = s & 63, r0 = (s >> 6) * 8;
      const size_t base = (size_t)(k0 + r0) * D_MODEL + col0 + n;
      const unsigned q0 = pk2(W[base], W[base + D_MODEL]);
      const unsigned q1 = pk2(W[base + 2 * D_MODEL], W[base + 3 * D_MODEL]);
      const unsigned q2 = pk2(W[base + 4 * D_MODEL], W[base + 5 * D_MODEL]);
      const unsigned q3 = pk2(W[base + 6 * D_MODEL], W[base + 7 * D_MODEL]);
      *(uint4*)&Blds[n][r0] = make_uint4(q0, q1, q2, q3);
    }
    if (k0 + BK < D_MODEL)
      __builtin_prefetch(&X[(size_t)(row0 + ar) * D_MODEL + k0 + BK + ac], 0, 0);
    __syncthreads();

    Frag16 a;
    const int m = (wave << 4) + l16;
    a.q[0] = *(const uint4*)&Alds[m][8 * h16];
    a.q[1] = *(const uint4*)&Alds[m][16 + 8 * h16];
#pragma unroll
    for (int t = 0; t < 4; ++t) {
      Frag16 b;
      const int n = t * 16 + l16;
      b.q[0] = *(const uint4*)&Blds[n][8 * h16];
      b.q[1] = *(const uint4*)&Blds[n][16 + 8 * h16];
      acc[t] = wmma_bf16(a, b, acc[t]);
    }
    __syncthreads();
  }

#pragma unroll
  for (int t = 0; t < 4; ++t) {
    const int n = col0 + t * 16 + l16;
    const float bv = bias[n];
    const int hh = n >> 6, dd = n & 63;
#pragma unroll
    for (int r = 0; r < 8; r += 2) {
      const int m = row0 + (wave << 4) + r + 8 * h16;
      const int bb = m >> 11, ss = m & 2047;
      const unsigned pp = pk2(acc[t][r] + bv, acc[t][r + 1] + bv);
      if (VT) {
        const size_t o = (((size_t)bb * NHEAD + hh) * DEPTH + dd) * SEQ + ss;
        out_hm[o] = (unsigned short)pp;
        out_hm[o + 1] = (unsigned short)(pp >> 16);
      } else {
        const size_t o = (((size_t)bb * NHEAD + hh) * SEQ + ss) * DEPTH + dd;
        out_hm[o] = (unsigned short)pp;
        out_hm[o + DEPTH] = (unsigned short)(pp >> 16);
      }
    }
  }
}

// ---------------------------------------------------------------------------
// Kernel 2: flash attention, one (b,h) per blockIdx.y, 64 q-rows per block.
// K [bh,s,d] and V^T [bh,d,s] double-buffered in LDS via async global->LDS.
// ---------------------------------------------------------------------------
#define QB 64
#define TB 32
#define NKB (SEQ / TB)

__global__ __launch_bounds__(128) void flash_attn_bf16(
    const unsigned short* __restrict__ Q, const unsigned short* __restrict__ K,
    const unsigned short* __restrict__ Vt, unsigned short* __restrict__ C) {
  __shared__ __align__(16) unsigned short Klds[2][TB][DEPTH + 8];
  __shared__ __align__(16) unsigned short Vlds[2][DEPTH][TB + 8]; // [d][t]
  __shared__ __align__(16) unsigned short Plds[4][16][TB + 8];
  const int tid = threadIdx.x, wave = tid >> 5, lane = tid & 31;
  const int h16 = lane >> 4, l16 = lane & 15;
  const int qb = blockIdx.x, bh = blockIdx.y;
  const size_t bh_off = (size_t)bh * SEQ * DEPTH;
  const unsigned short* Qp = Q + bh_off;
  const uint64_t kbase = (uint64_t)(uintptr_t)(K + bh_off);
  const uint64_t vbase = (uint64_t)(uintptr_t)(Vt + bh_off);

  // Q fragments for this wave's 16 rows (two depth halves).
  Frag16 qf[2];
  const unsigned short* qrow = Qp + (size_t)(qb * QB + wave * 16 + l16) * DEPTH;
  qf[0].q[0] = *(const uint4*)(qrow + 8 * h16);
  qf[0].q[1] = *(const uint4*)(qrow + 16 + 8 * h16);
  qf[1].q[0] = *(const uint4*)(qrow + 32 + 8 * h16);
  qf[1].q[1] = *(const uint4*)(qrow + 48 + 8 * h16);

  float mstate[8], lstate[8];
  v8f ctx[4] = {};
#pragma unroll
  for (int e = 0; e < 8; ++e) { mstate[e] = -1e30f; lstate[e] = 0.0f; }

  // Async copy geometry: K block 32x128B (2 chunks/thread), Vt block 64x64B
  // (1 chunk/thread); all b128.
  const int kr = tid >> 3, kc = tid & 7;
  const int vd = tid >> 1, vh = tid & 1;
  unsigned klA[2], klB[2], vl[2];
#pragma unroll
  for (int b = 0; b < 2; ++b) {
    klA[b] = (unsigned)(uintptr_t)&Klds[b][kr][kc * 8];
    klB[b] = (unsigned)(uintptr_t)&Klds[b][kr + 16][kc * 8];
    vl[b] = (unsigned)(uintptr_t)&Vlds[b][vd][vh * 8];
  }
  auto issue = [&](int buf, int kb) {
    const unsigned kg0 = (unsigned)((kb * TB + kr) * (DEPTH * 2) + kc * 16);
    const unsigned kg1 = kg0 + 16u * DEPTH * 2;
    const unsigned vg = (unsigned)(vd * (SEQ * 2) + kb * (TB * 2) + vh * 16);
    asm volatile(
        "global_load_async_to_lds_b128 %0, %3, %6\n\t"
        "global_load_async_to_lds_b128 %1, %4, %6\n\t"
        "global_load_async_to_lds_b128 %2, %5, %7"
        :
        : "v"(klA[buf]), "v"(klB[buf]), "v"(vl[buf]), "v"(kg0), "v"(kg1),
          "v"(vg), "s"(kbase), "s"(vbase)
        : "memory");
  };

  issue(0, 0);
  for (int kb = 0; kb < NKB; ++kb) {
    const int cur = kb & 1;
    if (kb + 1 < NKB) {
      issue(cur ^ 1, kb + 1); // overlap next block's copy with this compute
      asm volatile("s_wait_asynccnt 3" ::: "memory"); // current block landed
    } else {
      asm volatile("s_wait_asynccnt 0" ::: "memory");
    }
    __syncthreads();

    // S = Q(16x64) * K^T(64x32), fp32 accumulate (4 WMMAs).
    v8f sacc[2] = {};
#pragma unroll
    for (int nt = 0; nt < 2; ++nt) {
      const int t = nt * 16 + l16;
#pragma unroll
      for (int ks = 0; ks < 2; ++ks) {
        Frag16 bf;
        const unsigned short* krp = &Klds[cur][t][ks * 32 + 8 * h16];
        bf.q[0] = *(const uint4*)krp;
        bf.q[1] = *(const uint4*)(krp + 16);
        sacc[nt] = wmma_bf16(qf[ks], bf, sacc[nt]);
      }
    }
    sacc[0] *= 0.125f; // 1/sqrt(depth)
    sacc[1] *= 0.125f;

    // Online softmax; row r = e + 8*h16 spans a 16-lane half-group.
    float p0[8], p1[8], al[8];
#pragma unroll
    for (int e = 0; e < 8; ++e) {
      float rm = fmaxf(sacc[0][e], sacc[1][e]);
      rm = fmaxf(rm, __shfl_xor(rm, 1, 32));
      rm = fmaxf(rm, __shfl_xor(rm, 2, 32));
      rm = fmaxf(rm, __shfl_xor(rm, 4, 32));
      rm = fmaxf(rm, __shfl_xor(rm, 8, 32));
      const float nm = fmaxf(mstate[e], rm);
      al[e] = __expf(mstate[e] - nm);
      p0[e] = __expf(sacc[0][e] - nm);
      p1[e] = __expf(sacc[1][e] - nm);
      float rs = p0[e] + p1[e];
      rs += __shfl_xor(rs, 1, 32);
      rs += __shfl_xor(rs, 2, 32);
      rs += __shfl_xor(rs, 4, 32);
      rs += __shfl_xor(rs, 8, 32);
      lstate[e] = lstate[e] * al[e] + rs;
      mstate[e] = nm;
    }
    v8f av;
#pragma unroll
    for (int e = 0; e < 8; ++e) av[e] = al[e];
    ctx[0] *= av; ctx[1] *= av; ctx[2] *= av; ctx[3] *= av;

    // P: C/D layout -> A-fragment layout via per-wave LDS scratch.
    // Pack (p0,p1) with one v_cvt_pk; halves go to columns l16 and 16+l16.
#pragma unroll
    for (int e = 0; e < 8; ++e) {
      const unsigned pp = pk2(p0[e], p1[e]);
      Plds[wave][e + 8 * h16][l16] = (unsigned short)pp;
      Plds[wave][e + 8 * h16][16 + l16] = (unsigned short)(pp >> 16);
    }
    asm volatile("s_wait_dscnt 0" ::: "memory"); // DS in-order per wave
    Frag16 pf;
    pf.q[0] = *(const uint4*)&Plds[wave][l16][8 * h16];
    pf.q[1] = *(const uint4*)&Plds[wave][l16][16 + 8 * h16];

    // ctx += P(16x32) * V(32x64); V^T layout makes B-frags contiguous b128s.
#pragma unroll
    for (int nt = 0; nt < 4; ++nt) {
      Frag16 vf;
      const unsigned short* vr = &Vlds[cur][nt * 16 + l16][8 * h16];
      vf.q[0] = *(const uint4*)vr;
      vf.q[1] = *(const uint4*)(vr + 16);
      ctx[nt] = wmma_bf16(pf, vf, ctx[nt]);
    }
    __syncthreads();
  }

  // Normalize, store ctx bf16 in concat layout [B, S, H, DEPTH].
  const int b = bh >> 4, h = bh & 15;
  float rinv[8];
#pragma unroll
  for (int e = 0; e < 8; ++e) rinv[e] = 1.0f / lstate[e];
#pragma unroll
  for (int nt = 0; nt < 4; ++nt) {
    const int d = nt * 16 + l16;
#pragma unroll
    for (int e = 0; e < 8; e += 2) {
      const int s = qb * QB + wave * 16 + e + 8 * h16;
      const unsigned pp = pk2(ctx[nt][e] * rinv[e], ctx[nt][e + 1] * rinv[e + 1]);
      const size_t o = (((size_t)b * SEQ + s) * NHEAD + h) * DEPTH + d;
      C[o] = (unsigned short)pp;
      C[o + (size_t)NHEAD * DEPTH] = (unsigned short)(pp >> 16); // row s+1
    }
  }
}

// ---------------------------------------------------------------------------
// Kernel 3: out = ctx(bf16) * Wo + bo, fp32 output. A-tile streamed to LDS
// with async global->LDS (already bf16, no conversion needed).
// ---------------------------------------------------------------------------
__global__ __launch_bounds__(128) void out_gemm_f32(
    const unsigned short* __restrict__ A, const float* __restrict__ W,
    const float* __restrict__ bias, float* __restrict__ out) {
  __shared__ __align__(16) unsigned short Alds[BM][BK + 8];
  __shared__ __align__(16) unsigned short Blds[BN][BK + 8];
  const int tid = threadIdx.x, wave = tid >> 5, lane = tid & 31;
  const int h16 = lane >> 4, l16 = lane & 15;
  const int row0 = blockIdx.x * BM, col0 = blockIdx.y * BN;
  v8f acc[4] = {};

  const uint64_t abase = (uint64_t)(uintptr_t)A;
  const int ar = tid >> 2, ach = tid & 3; // 64 rows x 4 b128 chunks
  const unsigned al0 = (unsigned)(uintptr_t)&Alds[ar][ach * 8];
  const unsigned al1 = (unsigned)(uintptr_t)&Alds[ar + 32][ach * 8];

  for (int k0 = 0; k0 < D_MODEL; k0 += BK) {
    const unsigned ga0 =
        (unsigned)(((row0 + ar) * D_MODEL + k0) * 2 + ach * 16);
    const unsigned ga1 = ga0 + 32u * D_MODEL * 2;
    asm volatile(
        "global_load_async_to_lds_b128 %0, %2, %4\n\t"
        "global_load_async_to_lds_b128 %1, %3, %4"
        :
        : "v"(al0), "v"(al1), "v"(ga0), "v"(ga1), "s"(abase)
        : "memory");
#pragma unroll
    for (int s = tid; s < 4 * BN; s += 128) {
      const int n = s & 63, r0 = (s >> 6) * 8;
      const size_t base = (size_t)(k0 + r0) * D_MODEL + col0 + n;
      const unsigned q0 = pk2(W[base], W[base + D_MODEL]);
      const unsigned q1 = pk2(W[base + 2 * D_MODEL], W[base + 3 * D_MODEL]);
      const unsigned q2 = pk2(W[base + 4 * D_MODEL], W[base + 5 * D_MODEL]);
      const unsigned q3 = pk2(W[base + 6 * D_MODEL], W[base + 7 * D_MODEL]);
      *(uint4*)&Blds[n][r0] = make_uint4(q0, q1, q2, q3);
    }
    asm volatile("s_wait_asynccnt 0" ::: "memory");
    __syncthreads();

    Frag16 a;
    const int m = (wave << 4) + l16;
    a.q[0] = *(const uint4*)&Alds[m][8 * h16];
    a.q[1] = *(const uint4*)&Alds[m][16 + 8 * h16];
#pragma unroll
    for (int t = 0; t < 4; ++t) {
      Frag16 b;
      const int n = t * 16 + l16;
      b.q[0] = *(const uint4*)&Blds[n][8 * h16];
      b.q[1] = *(const uint4*)&Blds[n][16 + 8 * h16];
      acc[t] = wmma_bf16(a, b, acc[t]);
    }
    __syncthreads();
  }

#pragma unroll
  for (int t = 0; t < 4; ++t) {
    const int n = col0 + t * 16 + l16;
    const float bv = bias[n];
#pragma unroll
    for (int r = 0; r < 8; ++r) {
      const int m = row0 + (wave << 4) + r + 8 * h16;
      out[(size_t)m * D_MODEL + n] = acc[t][r] + bv;
    }
  }
}

// ---------------------------------------------------------------------------
extern "C" void kernel_launch(void* const* d_in, const int* in_sizes, int n_in,
                              void* d_out, int out_size, void* d_ws,
                              size_t ws_size, hipStream_t stream) {
  (void)in_sizes; (void)n_in; (void)out_size; (void)ws_size;
  const float* x = (const float*)d_in[0];
  const float* wq = (const float*)d_in[1];
  const float* bq = (const float*)d_in[2];
  const float* wk = (const float*)d_in[3];
  const float* bk = (const float*)d_in[4];
  const float* wv = (const float*)d_in[5];
  const float* bv = (const float*)d_in[6];
  const float* wo = (const float*)d_in[7];
  const float* bo = (const float*)d_in[8];

  unsigned short* ws = (unsigned short*)d_ws;
  const size_t N = (size_t)ROWS * D_MODEL; // 4 Mi elements each
  unsigned short* Qb = ws;
  unsigned short* Kb = ws + N;
  unsigned short* Vtb = ws + 2 * N; // [B,H,DEPTH,S]
  unsigned short* Cx = ws + 3 * N;

  dim3 gg(ROWS / BM, D_MODEL / BN);
  qkv_gemm_bf16<0><<<gg, 128, 0, stream>>>(x, wq, bq, Qb);
  qkv_gemm_bf16<0><<<gg, 128, 0, stream>>>(x, wk, bk, Kb);
  qkv_gemm_bf16<1><<<gg, 128, 0, stream>>>(x, wv, bv, Vtb);
  flash_attn_bf16<<<dim3(SEQ / QB, BATCH * NHEAD), 128, 0, stream>>>(Qb, Kb,
                                                                     Vtb, Cx);
  out_gemm_f32<<<gg, 128, 0, stream>>>(Cx, wo, bo, (float*)d_out);
}
